// Attention_12841952215634
// MI455X (gfx1250) — compile-verified, matrix operand order checked by reading, not run on hardware
//
#include <hip/hip_runtime.h>
#include <hip/hip_bf16.h>

typedef __attribute__((ext_vector_type(16))) _Float16 v16h;
typedef __attribute__((ext_vector_type(8)))  _Float16 v8h;
typedef __attribute__((ext_vector_type(8)))  float    v8f;
typedef __attribute__((ext_vector_type(4)))  int      v4i;

#define WMMA_F32_F16(A, B, C) \
    __builtin_amdgcn_wmma_f32_16x16x32_f16(false, (A), false, (B), (short)0, (C), false, false)

// ---------------------------------------------------------------------------
// CDNA5 async global->LDS staging (ASYNCcnt path), with sync fallback.
// Builtin signature (from hipcc diagnostic): first arg is a global-AS int4*.
// Generic LDS pointer low 32 bits == LDS byte offset (ISA 10.2 aperture map),
// so the 32-bit truncating cast to addrspace(3) is value-correct.
// ---------------------------------------------------------------------------
#ifndef __has_builtin
#define __has_builtin(x) 0
#endif
#if __has_builtin(__builtin_amdgcn_global_load_async_to_lds_b128) && \
    __has_builtin(__builtin_amdgcn_s_wait_asynccnt)
#define USE_ASYNC_LDS 1
#else
#define USE_ASYNC_LDS 0
#endif

#if USE_ASYNC_LDS
typedef __attribute__((address_space(1))) v4i as1_v4i;
typedef __attribute__((address_space(3))) v4i as3_v4i;
#define WAIT_ASYNC(n) __builtin_amdgcn_s_wait_asynccnt(n)
#else
#define WAIT_ASYNC(n)
#endif

__device__ __forceinline__ void copy_b128_g2l(void* lds_dst, const void* gsrc) {
#if USE_ASYNC_LDS
    __builtin_amdgcn_global_load_async_to_lds_b128(
        (as1_v4i*)(unsigned long long)gsrc,
        (as3_v4i*)(unsigned int)(unsigned long long)lds_dst,
        0, 0);
#else
    *(v8h*)lds_dst = *(const v8h*)gsrc;
#endif
}

// ---------------------------------------------------------------------------
// 16-bit A-matrix fragment loader (ISA 7.12.2). B fragments use the same
// pattern when the tile is staged as B^T row-major (row = N, col = K).
// ---------------------------------------------------------------------------
__device__ __forceinline__ v16h load_frag16(const _Float16* __restrict__ base,
                                            int row_stride, int lane) {
    const int r     = lane & 15;
    const int khalf = (lane >> 4) & 1;
    const _Float16* p = base + r * row_stride + khalf * 8;
    v8h lo = *(const v8h*)(p);        // K = 8*khalf .. +7
    v8h hi = *(const v8h*)(p + 16);   // K = 16+8*khalf .. +7
    v16h out;
#pragma unroll
    for (int i = 0; i < 8; ++i) { out[i] = lo[i]; out[8 + i] = hi[i]; }
    return out;
}

// ---------------------------------------------------------------------------
// fp32 -> f16 conversion
// ---------------------------------------------------------------------------
__global__ void cvt_f32_to_f16(const float* __restrict__ src,
                               _Float16* __restrict__ dst, int n) {
    int i = blockIdx.x * blockDim.x + threadIdx.x;
    if (i < n) dst[i] = (_Float16)src[i];
}

// ---------------------------------------------------------------------------
// C[M][Nout] = A[M][K] * W[Nout][K]^T (+ bias). f16 in, WMMA f32 acc.
// WG tile 128x256, 8 waves (2M x 4N), wave tile 64x64 (16 accumulators),
// BK=32, double-buffered LDS fed by async global->LDS copies.
// ---------------------------------------------------------------------------
template <bool OUT_F16>
__global__ void __launch_bounds__(256)
gemm_wmma(const _Float16* __restrict__ A,
          const _Float16* __restrict__ Wt,
          const float* __restrict__ bias,
          _Float16* __restrict__ Cf16,
          float* __restrict__ Cf32,
          int M, int Nout, int K) {
    __shared__ _Float16 As[2][128][40];
    __shared__ _Float16 Bs[2][256][40];

    const int tid  = threadIdx.x;
    const int wave = tid >> 5;
    const int lane = tid & 31;
    const int wm   = wave & 1;   // 0..1 -> 64-row slab
    const int wn   = wave >> 1;  // 0..3 -> 64-col slab
    const int tileM = blockIdx.y * 128;
    const int tileN = blockIdx.x * 256;

    v8f acc[4][4] = {};

    // Stage one BK=32 slice of A (128x32) + B (256x32): 6 b128 chunks/thread.
    auto stage = [&](int buf, int k0) {
#pragma unroll
        for (int j = 0; j < 2; ++j) {                 // A tile: rows 0..127
            int chunk = j * 256 + tid;
            int r = chunk >> 2;
            int c = (chunk & 3) * 8;
            copy_b128_g2l(&As[buf][r][c],
                          &A[(size_t)(tileM + r) * K + k0 + c]);
        }
#pragma unroll
        for (int j = 0; j < 4; ++j) {                 // B tile: rows 0..255
            int chunk = j * 256 + tid;
            int r = chunk >> 2;
            int c = (chunk & 3) * 8;
            copy_b128_g2l(&Bs[buf][r][c],
                          &Wt[(size_t)(tileN + r) * K + k0 + c]);
        }
    };

    const int nk = K >> 5;
    stage(0, 0);
    for (int kt = 0; kt < nk; ++kt) {
        const int cur = kt & 1;
        if (kt + 1 < nk) {
            stage(cur ^ 1, (kt + 1) << 5);  // prefetch next slice (async)
            WAIT_ASYNC(6);                  // oldest 6 (current slice) done
        } else {
            WAIT_ASYNC(0);
        }
        __syncthreads();

        v16h bfrag[4];
#pragma unroll
        for (int tn = 0; tn < 4; ++tn)
            bfrag[tn] = load_frag16(&Bs[cur][wn * 64 + tn * 16][0], 40, lane);
#pragma unroll
        for (int tm = 0; tm < 4; ++tm) {
            v16h afrag = load_frag16(&As[cur][wm * 64 + tm * 16][0], 40, lane);
#pragma unroll
            for (int tn = 0; tn < 4; ++tn)
                acc[tm][tn] = WMMA_F32_F16(afrag, bfrag[tn], acc[tm][tn]);
        }
        __syncthreads();
    }

    // Epilogue: C/D layout -> lane = col (lane&15), VGPR v -> row v+8*(lane>>4)
    const int half = lane >> 4;
    const int c0   = lane & 15;
#pragma unroll
    for (int tm = 0; tm < 4; ++tm) {
#pragma unroll
        for (int tn = 0; tn < 4; ++tn) {
#pragma unroll
            for (int v = 0; v < 8; ++v) {
                int row = tileM + wm * 64 + tm * 16 + v + 8 * half;
                int col = tileN + wn * 64 + tn * 16 + c0;
                float val = acc[tm][tn][v];
                if (OUT_F16) {
                    Cf16[(size_t)row * Nout + col] = (_Float16)val;
                } else {
                    Cf32[(size_t)row * Nout + col] = val + bias[col];
                }
            }
        }
    }
}

// ---------------------------------------------------------------------------
// Flash attention: grid (B*H, N/128); 8 waves, 16 queries per wave.
// qkv layout: [B*N][2304], e = s*768 + h*64 + d (s = 0:Q 1:K 2:V).
// Output O: [B*N][768] f16, col = h*64 + d.
// ---------------------------------------------------------------------------
__global__ void __launch_bounds__(256)
attn_wmma(const _Float16* __restrict__ qkv, _Float16* __restrict__ O) {
    const int bh = blockIdx.x;             // 0..191
    const int b = bh / 12, h = bh % 12;
    const int qblk = blockIdx.y;           // 0..7

    __shared__ _Float16 Ks[32][72];        // 32 keys x 64 d   (B^T for QK^T)
    __shared__ _Float16 Vt[64][40];        // 64 d x 32 keys   (B^T for P.V)
    __shared__ _Float16 Ps[8][16][40];     // per-wave P transpose buffer

    const int tid = threadIdx.x, wave = tid >> 5, lane = tid & 31;
    const int rowbase = b * 1024;
    const int qrow0   = rowbase + qblk * 128 + wave * 16;
    const int qcol    = h * 64;
    const float SCALE = 0.125f;            // 1/sqrt(64)

    // Q fragments (16 q x 64 d -> two 16x32 A fragments), kept in VGPRs.
    v16h qf[2];
    {
        const int r = lane & 15, khalf = (lane >> 4) & 1;
        const _Float16* qp = qkv + (size_t)(qrow0 + r) * 2304 + qcol;
#pragma unroll
        for (int f = 0; f < 2; ++f) {
            const _Float16* p = qp + f * 32 + khalf * 8;
            v8h lo = *(const v8h*)p;
            v8h hi = *(const v8h*)(p + 16);
#pragma unroll
            for (int i = 0; i < 8; ++i) { qf[f][i] = lo[i]; qf[f][8 + i] = hi[i]; }
        }
    }

    v8f acc_o[4] = {};
    float m[8], l[8];
#pragma unroll
    for (int v = 0; v < 8; ++v) { m[v] = -1e30f; l[v] = 0.0f; }

    for (int kk = 0; kk < 1024; kk += 32) {
        // K tile: async global->LDS. V tile: sync load + in-flight transpose.
        {
            const int r = tid >> 3;            // 0..31 key row
            const int c = (tid & 7) * 8;       // 0..56 d offset
            copy_b128_g2l(&Ks[r][c],
                          qkv + (size_t)(rowbase + kk + r) * 2304 + 768 + qcol + c);
            const _Float16* vp =
                qkv + (size_t)(rowbase + kk + r) * 2304 + 1536 + qcol + c;
            v8h vv = *(const v8h*)vp;
#pragma unroll
            for (int i = 0; i < 8; ++i) Vt[c + i][r] = vv[i];
        }
        WAIT_ASYNC(0);
        __syncthreads();

        // S = Q K^T : two 16-key N tiles, K-dim 64 in two 32 steps.
        v8f s0 = {}, s1 = {};
#pragma unroll
        for (int dstep = 0; dstep < 2; ++dstep) {
            v16h kf0 = load_frag16(&Ks[0][dstep * 32], 72, lane);
            v16h kf1 = load_frag16(&Ks[16][dstep * 32], 72, lane);
            s0 = WMMA_F32_F16(qf[dstep], kf0, s0);
            s1 = WMMA_F32_F16(qf[dstep], kf1, s1);
        }

        // Online softmax (f32). Row stats per half-wave, per VGPR slot.
#pragma unroll
        for (int v = 0; v < 8; ++v) {
            float a0 = s0[v] * SCALE;
            float a1 = s1[v] * SCALE;
            float mx = fmaxf(a0, a1);
#pragma unroll
            for (int off = 1; off < 16; off <<= 1)
                mx = fmaxf(mx, __shfl_xor(mx, off, 32));
            float nm   = fmaxf(m[v], mx);
            float corr = __expf(m[v] - nm);
            float p0   = __expf(a0 - nm);
            float p1   = __expf(a1 - nm);
            float ps   = p0 + p1;
#pragma unroll
            for (int off = 1; off < 16; off <<= 1)
                ps += __shfl_xor(ps, off, 32);
            l[v] = l[v] * corr + ps;
            m[v] = nm;
#pragma unroll
            for (int t = 0; t < 4; ++t) acc_o[t][v] *= corr;
            // Transpose P through LDS: C layout (lane=col) -> A layout (lane=row).
            int r = v + 8 * (lane >> 4);
            int c = lane & 15;
            Ps[wave][r][c]      = (_Float16)p0;
            Ps[wave][r][c + 16] = (_Float16)p1;
        }

        // O += P * V (4 d-tiles). Same-wave LDS ops are in-order (DScnt).
        v16h pf = load_frag16(&Ps[wave][0][0], 40, lane);
#pragma unroll
        for (int t = 0; t < 4; ++t) {
            v16h vf = load_frag16(&Vt[t * 16][0], 40, lane);
            acc_o[t] = WMMA_F32_F16(pf, vf, acc_o[t]);
        }
        __syncthreads();
    }

    // Normalize and write O (head-major [B*N][768] f16).
    const int half = lane >> 4, c0 = lane & 15;
#pragma unroll
    for (int t = 0; t < 4; ++t) {
#pragma unroll
        for (int v = 0; v < 8; ++v) {
            float ov = acc_o[t][v] / l[v];
            int row = qrow0 + v + 8 * half;
            O[(size_t)row * 768 + qcol + t * 16 + c0] = (_Float16)ov;
        }
    }
}

// ---------------------------------------------------------------------------
// Host side
// ---------------------------------------------------------------------------
static inline size_t align256(size_t x) { return (x + 255) & ~(size_t)255; }

extern "C" void kernel_launch(void* const* d_in, const int* in_sizes, int n_in,
                              void* d_out, int out_size, void* d_ws, size_t ws_size,
                              hipStream_t stream) {
    (void)in_sizes; (void)n_in; (void)out_size; (void)ws_size;

    const float* x      = (const float*)d_in[0];   // [16,1024,768]
    const float* w_qkv  = (const float*)d_in[1];   // [2304,768]
    const float* w_proj = (const float*)d_in[2];   // [768,768]
    const float* b_proj = (const float*)d_in[3];   // [768]
    float* out = (float*)d_out;

    const int Bv = 16, Nv = 1024, DIM = 768;
    const int M = Bv * Nv;                 // 16384
    const int E = 3 * DIM;                 // 2304
    const size_t nX   = (size_t)M * DIM;
    const size_t nWq  = (size_t)E * DIM;
    const size_t nWp  = (size_t)DIM * DIM;
    const size_t nQKV = (size_t)M * E;

    char* ws = (char*)d_ws;
    size_t off = 0;
    _Float16* xh   = (_Float16*)(ws + off); off = align256(off + nX   * 2);
    _Float16* wqh  = (_Float16*)(ws + off); off = align256(off + nWq  * 2);
    _Float16* wph  = (_Float16*)(ws + off); off = align256(off + nWp  * 2);
    _Float16* qkvh = (_Float16*)(ws + off); off = align256(off + nQKV * 2);
    _Float16* oh   = (_Float16*)(ws + off); off = align256(off + nX   * 2);

    // 1) fp32 -> f16 conversions
    cvt_f32_to_f16<<<(int)((nX  + 255) / 256), 256, 0, stream>>>(x,      xh,  (int)nX);
    cvt_f32_to_f16<<<(int)((nWq + 255) / 256), 256, 0, stream>>>(w_qkv,  wqh, (int)nWq);
    cvt_f32_to_f16<<<(int)((nWp + 255) / 256), 256, 0, stream>>>(w_proj, wph, (int)nWp);

    // 2) QKV projection: [16384,2304] = xh * wqh^T
    gemm_wmma<true><<<dim3(E / 256, M / 128), 256, 0, stream>>>(
        xh, wqh, nullptr, qkvh, nullptr, M, E, DIM);

    // 3) Attention: grid (B*H, N/128)
    attn_wmma<<<dim3(Bv * 12, Nv / 128), 256, 0, stream>>>(qkvh, oh);

    // 4) Output projection + bias: out = oh * wph^T + b
    gemm_wmma<false><<<dim3(DIM / 256, M / 128), 256, 0, stream>>>(
        oh, wph, b_proj, nullptr, out, M, DIM, DIM);
}
